// Attention_global_46067819217520
// MI455X (gfx1250) — matrix operand order repaired and checked by reading, not compiled, not run on hardware
//
#include <hip/hip_runtime.h>
#include <hip/hip_bf16.h>

#define B_   16
#define L_   1024
#define DIM_ 256
#define H_   8
#define DH_  32
#define QK_  64

#define USE_ASYNC_LDS 1

typedef __attribute__((ext_vector_type(16))) __bf16        v16bf;
typedef __attribute__((ext_vector_type(8)))  float         v8f;
typedef __attribute__((ext_vector_type(4)))  unsigned int  v4u;

// ---------- bf16 helpers (bit-level, RNE) ----------
__device__ __forceinline__ unsigned short f2bfu(float f) {
    union { float f; unsigned u; } c; c.f = f;
    unsigned u = c.u;
    unsigned r = u + 0x7FFFu + ((u >> 16) & 1u);
    return (unsigned short)(r >> 16);
}

// ---------- CDNA5 async global->LDS copy (ASYNCcnt-tracked) ----------
__device__ __forceinline__ void async_g2l_b128(void* lds_ptr, const void* gptr) {
    unsigned int       loff = (unsigned int)(uintptr_t)lds_ptr;   // low 32b = LDS offset
    unsigned long long ga   = (unsigned long long)(uintptr_t)gptr;
    asm volatile("global_load_async_to_lds_b128 %0, %1, off"
                 :: "v"(loff), "v"(ga) : "memory");
}
__device__ __forceinline__ void wait_async0() {
    asm volatile("s_wait_asynccnt 0x0" ::: "memory");
}

// ---------- WMMA fragment loaders (ISA 7.12.2 layouts, wave32) ----------
// A 16x32 bf16: lane = 16*half + m. Elements for VGPR v are K=k0,k0+1 with
// k0 = (v<4 ? 2v : 8+2v) + 8*half  ==> two contiguous 16B runs per lane:
//   shorts [8*half .. 8*half+7] and [8*half+16 .. 8*half+23]
__device__ __forceinline__ v16bf load_a_frag(const unsigned short* base, int rstride) {
    int lane = threadIdx.x & 31;
    int half = lane >> 4;
    int m    = lane & 15;
    const unsigned short* p = base + (size_t)m * rstride + 8 * half;
    union { v4u q[2]; v16bf v; } r;
    r.q[0] = *(const v4u*)(p);
    r.q[1] = *(const v4u*)(p + 16);
    return r.v;
}
// B 32x16 bf16 from column-major storage: element B[k][n] = base[n*cstride + k].
// Lane (half,n) needs K = 8*half..8*half+7 and 8*half+16..8*half+23 (same as A).
__device__ __forceinline__ v16bf load_b_frag_t(const unsigned short* base, int cstride) {
    int lane = threadIdx.x & 31;
    int half = lane >> 4;
    int n    = lane & 15;
    const unsigned short* p = base + (size_t)n * cstride + 8 * half;
    union { v4u q[2]; v16bf v; } r;
    r.q[0] = *(const v4u*)(p);
    r.q[1] = *(const v4u*)(p + 16);
    return r.v;
}

__device__ __forceinline__ v8f wmma_bf16(v16bf a, v16bf b, v8f c) {
    return __builtin_amdgcn_wmma_f32_16x16x32_bf16(false, a, false, b, (short)0, c, false, false);
}

// reductions within a 16-lane half (rows of C live per-half)
__device__ __forceinline__ float half16_max(float v) {
    v = fmaxf(v, __shfl_xor(v, 1, 32));
    v = fmaxf(v, __shfl_xor(v, 2, 32));
    v = fmaxf(v, __shfl_xor(v, 4, 32));
    v = fmaxf(v, __shfl_xor(v, 8, 32));
    return v;
}
__device__ __forceinline__ float half16_sum(float v) {
    v += __shfl_xor(v, 1, 32);
    v += __shfl_xor(v, 2, 32);
    v += __shfl_xor(v, 4, 32);
    v += __shfl_xor(v, 8, 32);
    return v;
}

__device__ __forceinline__ float gelu_tanh(float x) {
    float x3 = x * x * x;
    float t  = tanhf(0.7978845608028654f * (x + 0.044715f * x3));
    return 0.5f * x * (1.0f + t);
}

// ============ kernel 0: tiled transpose + fp32->bf16 =============
// Produces x^T [B][DIM][L] and Wproj^T [DIM out][DIM in] in bf16 so that all
// WMMA B-fragments are two contiguous 16B runs per lane.
__global__ __launch_bounds__(256) void k_transpose_bf(
    const float* __restrict__ x, const float* __restrict__ wproj,
    unsigned short* __restrict__ xT, unsigned short* __restrict__ wpT) {
    __shared__ __align__(16) float s_t[32][33];
    const int bid = blockIdx.x;
    const float* src;
    unsigned short* dst;
    int srcStride, dstStride;
    if (bid < 4096) {                       // 16 b * 32 t-tiles * 8 c-tiles
        int b  = bid >> 8;
        int t  = bid & 255;
        int tt = t >> 3;                    // t-tile (rows of src)
        int ct = t & 7;                     // dim-tile (cols of src)
        src = x + ((size_t)b * L_ + tt * 32) * DIM_ + ct * 32;
        srcStride = DIM_;
        dst = xT + ((size_t)b * DIM_ + ct * 32) * L_ + tt * 32;
        dstStride = L_;
    } else {                                // 8x8 tiles of W_proj
        int t  = bid - 4096;
        int kt = t >> 3, nt = t & 7;
        src = wproj + (size_t)(kt * 32) * DIM_ + nt * 32;
        srcStride = DIM_;
        dst = wpT + (size_t)(nt * 32) * DIM_ + kt * 32;
        dstStride = DIM_;
    }
    const int tid = threadIdx.x;
    {
        int r = tid >> 3, c4 = (tid & 7) * 4;
        const float4 v = *(const float4*)(src + (size_t)r * srcStride + c4);
        s_t[r][c4 + 0] = v.x; s_t[r][c4 + 1] = v.y;
        s_t[r][c4 + 2] = v.z; s_t[r][c4 + 3] = v.w;
    }
    __syncthreads();
    {
        int c = tid >> 3, r4 = (tid & 7) * 4;
        uint2 pk;
        pk.x = f2bfu(s_t[r4 + 0][c]) | ((unsigned)f2bfu(s_t[r4 + 1][c]) << 16);
        pk.y = f2bfu(s_t[r4 + 2][c]) | ((unsigned)f2bfu(s_t[r4 + 3][c]) << 16);
        *(uint2*)(dst + (size_t)c * dstStride + r4) = pk;
    }
}

// ================= kernel 1: qk GIN =================
// per block: (b, 32 rows).  agg = adj @ x via bf16 WMMA (A=adj tile, B=x^T
// tile), h1 = (1+eps)x + agg, f32 MLP 32->64->64 per head -> q,k (bf16).
__global__ __launch_bounds__(256) void k_qk_gin(
    const float* __restrict__ x, const float* __restrict__ adj,
    const float* __restrict__ eps_qk,
    const float* __restrict__ W1, const float* __restrict__ b1,
    const float* __restrict__ W2, const float* __restrict__ b2,
    const unsigned short* __restrict__ xT,
    unsigned short* __restrict__ qbf, unsigned short* __restrict__ kbf) {
    // phase1: adj tile (2KB) + x^T tile (20KB); phase2: h1 (32KB) reuses region
    __shared__ __align__(16) unsigned char s_dyn[32768];
    __shared__ float s_W1[DH_ * QK_];
    __shared__ float s_W2[QK_ * QK_];
    __shared__ float s_b1[QK_];
    __shared__ float s_b2[QK_];

    unsigned short* s_adj = (unsigned short*)s_dyn;            // [32][32]
    unsigned short* s_xT  = (unsigned short*)(s_dyn + 2048);   // [256 cols][40]
    float*          s_h1  = (float*)s_dyn;                     // [32][256]

    const int tid  = threadIdx.x;
    const int lane = tid & 31;
    const int wv   = tid >> 5;
    const int b    = blockIdx.y;
    const int l0   = blockIdx.x * 32;

    for (int i = tid; i < DH_ * QK_; i += 256) s_W1[i] = W1[i];
    for (int i = tid; i < QK_ * QK_; i += 256) s_W2[i] = W2[i];
    if (tid < QK_) { s_b1[tid] = b1[tid]; s_b2[tid] = b2[tid]; }

    const int mt = wv & 1, m0 = mt * 16;
    const int grp = wv >> 1;

    v8f acc[4];
#pragma unroll
    for (int j = 0; j < 4; ++j)
#pragma unroll
        for (int v = 0; v < 8; ++v) acc[j][v] = 0.0f;

    const float*          adj_row = adj + ((size_t)b * L_ + l0) * L_;
    const unsigned short* xT_b    = xT + (size_t)b * DIM_ * L_;

    for (int kc = 0; kc < 32; ++kc) {
        __syncthreads();
        {   // adj tile [32 r][32 k]: one float4 per thread -> packed 8B LDS store
            int r = tid >> 3, c = (tid & 7) * 4;
            const float4 a4 = *(const float4*)(adj_row + (size_t)r * L_ + kc * 32 + c);
            uint2 pk;
            pk.x = f2bfu(a4.x) | ((unsigned)f2bfu(a4.y) << 16);
            pk.y = f2bfu(a4.z) | ((unsigned)f2bfu(a4.w) << 16);
            *(uint2*)(s_adj + r * 32 + c) = pk;
        }
        // x^T tile [256 cols][32 k], col stride 40 shorts (16B aligned)
#if USE_ASYNC_LDS
#pragma unroll
        for (int k2 = 0; k2 < 4; ++k2) {
            int idx = tid + k2 * 256;
            int c = idx >> 2, part = idx & 3;
            async_g2l_b128((char*)s_xT + (size_t)c * 80 + part * 16,
                           xT_b + (size_t)c * L_ + kc * 32 + part * 8);
        }
        wait_async0();
#else
#pragma unroll
        for (int k2 = 0; k2 < 4; ++k2) {
            int idx = tid + k2 * 256;
            int c = idx >> 2, part = idx & 3;
            *(v4u*)((char*)s_xT + (size_t)c * 80 + part * 16) =
                *(const v4u*)(xT_b + (size_t)c * L_ + kc * 32 + part * 8);
        }
#endif
        __syncthreads();
        v16bf af = load_a_frag(s_adj + m0 * 32, 32);
#pragma unroll
        for (int j = 0; j < 4; ++j) {
            int   n0 = (grp * 4 + j) * 16;
            v16bf bf = load_b_frag_t(s_xT + n0 * 40, 40);
            acc[j] = wmma_bf16(af, bf, acc[j]);
        }
    }
    __syncthreads();
    // write agg into h1 region (reuses tile LDS)
    {
        int half = lane >> 4, n = lane & 15;
#pragma unroll
        for (int j = 0; j < 4; ++j) {
            int n0 = (grp * 4 + j) * 16;
#pragma unroll
            for (int v = 0; v < 8; ++v)
                s_h1[(m0 + v + 8 * half) * DIM_ + n0 + n] = acc[j][v];
        }
    }
    __syncthreads();
    const float epf = 1.0f + eps_qk[0];
    for (int i = tid; i < 32 * DIM_; i += 256) {
        int r = i >> 8, c = i & 255;
        s_h1[i] += epf * x[((size_t)b * L_ + l0 + r) * DIM_ + c];
    }
    __syncthreads();
    // MLP per (head, row): 256 threads = 8 heads x 32 rows
    {
        int h = tid >> 5, row = tid & 31;
        float hv[DH_];
#pragma unroll
        for (int d = 0; d < DH_; ++d) hv[d] = s_h1[row * DIM_ + h * DH_ + d];
        float hid[QK_];
        for (int j = 0; j < QK_; ++j) {
            float s = s_b1[j];
#pragma unroll
            for (int d = 0; d < DH_; ++d) s = fmaf(hv[d], s_W1[d * QK_ + j], s);
            hid[j] = fmaxf(s, 0.0f);
        }
        size_t obase = (((size_t)b * H_ + h) * L_ + l0 + row) * DH_;
        for (int o = 0; o < QK_; ++o) {
            float s = s_b2[o];
#pragma unroll
            for (int j = 0; j < QK_; ++j) s = fmaf(hid[j], s_W2[j * QK_ + o], s);
            if (o < DH_) qbf[obase + o]       = f2bfu(s);
            else         kbf[obase + o - DH_] = f2bfu(s);
        }
    }
}

// ================= kernel 2: fused attention + v GIN =================
// per block: (b, h, 32 rows). Pass1: q@k^T online softmax stats (DH=32 = one
// WMMA K-step). Pass2: recompute scores, write norm_attn once, push P through
// LDS as A-frags for P @ x (B from x^T, contiguous loads).
__global__ __launch_bounds__(256) void k_attn(
    const float* __restrict__ x,
    const float* __restrict__ eps_v,
    const float* __restrict__ W1, const float* __restrict__ b1,
    const float* __restrict__ W2, const float* __restrict__ b2,
    const unsigned short* __restrict__ xT,
    const unsigned short* __restrict__ qbf, const unsigned short* __restrict__ kbf,
    float* __restrict__ attn_out, float* __restrict__ v_ws) {
    __shared__ __align__(16) unsigned short s_q[32 * 32];
    __shared__ __align__(16) unsigned short s_P[8][16][40];
    __shared__ float s_m[2][16][4];
    __shared__ float s_s[2][16][4];
    __shared__ float s_agg[32 * DH_];
    __shared__ float s_W1[DH_ * DH_], s_W2[DH_ * DH_], s_b1v[DH_], s_b2v[DH_];

    const int tid = threadIdx.x, lane = tid & 31, wv = tid >> 5;
    const int half = lane >> 4, nn = lane & 15;
    const int b = blockIdx.z, h = blockIdx.y, l0 = blockIdx.x * 32;

    for (int i = tid; i < DH_ * DH_; i += 256) { s_W1[i] = W1[i]; s_W2[i] = W2[i]; }
    if (tid < DH_) { s_b1v[tid] = b1[tid]; s_b2v[tid] = b2[tid]; }
    for (int i = tid; i < 32 * DH_; i += 256) s_agg[i] = 0.0f;

    const size_t bh = (size_t)b * H_ + h;
    const unsigned short* qsrc = qbf + (bh * L_ + l0) * DH_;
#if USE_ASYNC_LDS
    if (tid < 128)   // 2KB q tile, wave-uniform split
        async_g2l_b128((char*)s_q + tid * 16, qsrc + tid * 8);
    wait_async0();
#else
    {
        const v4u* src = (const v4u*)qsrc;
        v4u*       dst = (v4u*)s_q;
        if (tid < 128) dst[tid] = src[tid];
    }
#endif
    __syncthreads();

    const int mt = wv & 1, m0 = mt * 16, grp = wv >> 1;
    v16bf aq = load_a_frag(s_q + m0 * 32, 32);
    const unsigned short* kbase = kbf + bh * L_ * DH_;

    const float scale = 0.03125f;  // 256^-0.5 / 2  (softmax temp folded in)
    float mrow[8], srow[8];
#pragma unroll
    for (int v = 0; v < 8; ++v) { mrow[v] = -1e30f; srow[v] = 0.0f; }

    // ---- pass 1: online max/sum over this wave's 8 K-chunks ----
    for (int i = 0; i < 8; ++i) {
        int t0 = (grp + 4 * i) * 32;
#pragma unroll
        for (int sub = 0; sub < 2; ++sub) {
            v16bf bk = load_b_frag_t(kbase + (size_t)(t0 + 16 * sub) * DH_, DH_);
            v8f   sc;
#pragma unroll
            for (int v = 0; v < 8; ++v) sc[v] = 0.0f;
            sc = wmma_bf16(aq, bk, sc);
#pragma unroll
            for (int v = 0; v < 8; ++v) {
                float s  = sc[v] * scale;
                float tm = half16_max(s);
                float e  = __expf(s - tm);
                float ts = half16_sum(e);
                float nm = fmaxf(mrow[v], tm);
                srow[v]  = srow[v] * __expf(mrow[v] - nm) + ts * __expf(tm - nm);
                mrow[v]  = nm;
            }
        }
    }
    if (nn == 0) {
#pragma unroll
        for (int v = 0; v < 8; ++v) {
            int r = v + 8 * half;
            s_m[mt][r][grp] = mrow[v];
            s_s[mt][r][grp] = srow[v];
        }
    }
    __syncthreads();
    // merge stats across the 4 waves sharing this row tile
    float Mv[8], Sinv[8];
#pragma unroll
    for (int v = 0; v < 8; ++v) {
        int   r = v + 8 * half;
        float M = fmaxf(fmaxf(s_m[mt][r][0], s_m[mt][r][1]),
                        fmaxf(s_m[mt][r][2], s_m[mt][r][3]));
        float S = s_s[mt][r][0] * __expf(s_m[mt][r][0] - M)
                + s_s[mt][r][1] * __expf(s_m[mt][r][1] - M)
                + s_s[mt][r][2] * __expf(s_m[mt][r][2] - M)
                + s_s[mt][r][3] * __expf(s_m[mt][r][3] - M);
        Mv[v]   = M;
        Sinv[v] = 1.0f / S;
    }

    // ---- pass 2: recompute, write norm_attn, accumulate P @ x ----
    v8f accv[2];
#pragma unroll
    for (int t = 0; t < 2; ++t)
#pragma unroll
        for (int v = 0; v < 8; ++v) accv[t][v] = 0.0f;

    const unsigned short* xTb = xT + ((size_t)b * DIM_ + h * DH_) * L_;
    for (int i = 0; i < 8; ++i) {
        int t0 = (grp + 4 * i) * 32;
#pragma unroll
        for (int sub = 0; sub < 2; ++sub) {
            v16bf bk = load_b_frag_t(kbase + (size_t)(t0 + 16 * sub) * DH_, DH_);
            v8f   sc;
#pragma unroll
            for (int v = 0; v < 8; ++v) sc[v] = 0.0f;
            sc = wmma_bf16(aq, bk, sc);
#pragma unroll
            for (int v = 0; v < 8; ++v) {
                float p = __expf(sc[v] * scale - Mv[v]) * Sinv[v];
                int   r = v + 8 * half;
                attn_out[(bh * L_ + (size_t)(l0 + m0 + r)) * L_ + t0 + 16 * sub + nn] = p;
                s_P[wv][r][16 * sub + nn] = f2bfu(p);
            }
        }
        // P tile [16 rows][32 k] is wave-private in LDS (DS ops in-order)
        v16bf ap = load_a_frag(&s_P[wv][0][0], 40);
#pragma unroll
        for (int t = 0; t < 2; ++t) {
            v16bf bx = load_b_frag_t(xTb + (size_t)(t * 16) * L_ + t0, L_);
            accv[t] = wmma_bf16(ap, bx, accv[t]);
        }
    }
    // cross-wave reduce agg (ds_add_f32)
#pragma unroll
    for (int t = 0; t < 2; ++t)
#pragma unroll
        for (int v = 0; v < 8; ++v)
            atomicAdd(&s_agg[(m0 + v + 8 * half) * DH_ + t * 16 + nn], accv[t][v]);
    __syncthreads();

    // v GIN MLP (32 rows, small): h1 = (1+eps)x + agg; relu(h1@W1+b1)@W2+b2
    if (tid < 32) {
        int         row = tid;
        const float epf = 1.0f + eps_v[0];
        float       hv[DH_];
#pragma unroll
        for (int d = 0; d < DH_; ++d)
            hv[d] = epf * x[((size_t)b * L_ + l0 + row) * DIM_ + h * DH_ + d]
                  + s_agg[row * DH_ + d];
        float hid[DH_];
        for (int j = 0; j < DH_; ++j) {
            float s = s_b1v[j];
#pragma unroll
            for (int d = 0; d < DH_; ++d) s = fmaf(hv[d], s_W1[d * DH_ + j], s);
            hid[j] = fmaxf(s, 0.0f);
        }
        for (int o = 0; o < DH_; ++o) {
            float s = s_b2v[o];
#pragma unroll
            for (int j = 0; j < DH_; ++j) s = fmaf(hid[j], s_W2[j * DH_ + o], s);
            // store head-transposed: v[b, l, h*DH + o]
            v_ws[((size_t)b * L_ + l0 + row) * DIM_ + h * DH_ + o] = s;
        }
    }
}

// ================= kernel 3: out = gelu(v) @ W_proj =================
__global__ __launch_bounds__(256) void k_proj(
    const float* __restrict__ v_ws, const unsigned short* __restrict__ wpT,
    float* __restrict__ out) {
    __shared__ __align__(16) unsigned short s_A[64 * 40];
    const int    tid = threadIdx.x, lane = tid & 31, wv = tid >> 5;
    const int    half = lane >> 4, nn = lane & 15;
    const size_t r0 = (size_t)blockIdx.x * 64;
    const int    mt = wv & 3, m0 = mt * 16, grp = wv >> 2;

    v8f acc[8];
#pragma unroll
    for (int j = 0; j < 8; ++j)
#pragma unroll
        for (int v = 0; v < 8; ++v) acc[j][v] = 0.0f;

    for (int kc = 0; kc < 8; ++kc) {
        __syncthreads();
#pragma unroll
        for (int k2 = 0; k2 < 4; ++k2) {      // 64x32 tile, float2 -> packed b32
            int idx = tid + k2 * 256;
            int r = idx >> 4, c2 = (idx & 15) * 2;
            const float2 v2 = *(const float2*)(v_ws + (size_t)(r0 + r) * DIM_ + kc * 32 + c2);
            unsigned int pk = f2bfu(gelu_tanh(v2.x)) |
                              ((unsigned)f2bfu(gelu_tanh(v2.y)) << 16);
            *(unsigned int*)(s_A + r * 40 + c2) = pk;
        }
        __syncthreads();
        v16bf af = load_a_frag(s_A + m0 * 40, 40);
#pragma unroll
        for (int j = 0; j < 8; ++j) {
            int   n0 = (grp * 8 + j) * 16;
            v16bf bf = load_b_frag_t(wpT + (size_t)n0 * DIM_ + kc * 32, DIM_);
            acc[j] = wmma_bf16(af, bf, acc[j]);
        }
    }
#pragma unroll
    for (int j = 0; j < 8; ++j) {
        int n0 = (grp * 8 + j) * 16;
#pragma unroll
        for (int v = 0; v < 8; ++v)
            out[(r0 + m0 + v + 8 * half) * DIM_ + n0 + nn] = acc[j][v];
    }
}

// ================= host launcher =================
extern "C" void kernel_launch(void* const* d_in, const int* in_sizes, int n_in,
                              void* d_out, int out_size, void* d_ws, size_t ws_size,
                              hipStream_t stream) {
    const float* x      = (const float*)d_in[0];
    const float* adj    = (const float*)d_in[1];
    // d_in[2] rep_adj_dis: unused by the reference
    const float* eps_qk = (const float*)d_in[3];
    const float* W1_qk  = (const float*)d_in[4];
    const float* b1_qk  = (const float*)d_in[5];
    const float* W2_qk  = (const float*)d_in[6];
    const float* b2_qk  = (const float*)d_in[7];
    const float* eps_v  = (const float*)d_in[8];
    const float* W1_v   = (const float*)d_in[9];
    const float* b1_v   = (const float*)d_in[10];
    const float* W2_v   = (const float*)d_in[11];
    const float* b2_v   = (const float*)d_in[12];
    const float* W_proj = (const float*)d_in[13];

    unsigned char*  ws   = (unsigned char*)d_ws;
    unsigned short* xT   = (unsigned short*)(ws);                       // 8 MB  x^T bf16
    unsigned short* qbf  = (unsigned short*)(ws + ((size_t)8  << 20));  // 8 MB
    unsigned short* kbf  = (unsigned short*)(ws + ((size_t)16 << 20));  // 8 MB
    float*          v_ws = (float*)(ws + ((size_t)24 << 20));           // 16 MB
    unsigned short* wpT  = (unsigned short*)(ws + ((size_t)40 << 20));  // 128 KB

    float* out      = (float*)d_out;
    float* attn_out = out + (size_t)B_ * L_ * DIM_;

    k_transpose_bf<<<dim3(4096 + 64), 256, 0, stream>>>(x, W_proj, xT, wpT);
    k_qk_gin<<<dim3(L_ / 32, B_), 256, 0, stream>>>(
        x, adj, eps_qk, W1_qk, b1_qk, W2_qk, b2_qk, xT, qbf, kbf);
    k_attn<<<dim3(L_ / 32, H_, B_), 256, 0, stream>>>(
        x, eps_v, W1_v, b1_v, W2_v, b2_v, xT, qbf, kbf, attn_out, v_ws);
    k_proj<<<dim3((B_ * L_) / 64), 256, 0, stream>>>(v_ws, wpT, out);
}
